// CausalSelfAttention_30520037606061
// MI455X (gfx1250) — compile-verified, matrix operand order checked by reading, not run on hardware
//
#include <hip/hip_runtime.h>
#include <math.h>

typedef __attribute__((ext_vector_type(16))) __bf16 v16bf;
typedef __attribute__((ext_vector_type(8)))  __bf16 v8bf;
typedef __attribute__((ext_vector_type(2)))  __bf16 v2bf;
typedef __attribute__((ext_vector_type(8)))  float  v8f;

#define T_LEN 4096
#define EDIM 1024
#define NH 8
#define HDIM 128
#define ATTN_SCALE 0.12f

__device__ __forceinline__ v8f wmma_bf16(v16bf a, v16bf b, v8f c) {
  return __builtin_amdgcn_wmma_f32_16x16x32_bf16(false, a, false, b, (short)0, c, false, false);
}

// Fragment loaders. Per ISA 7.12.2, a lane's 16 elements of a 16-bit A/B
// fragment are two contiguous 8-element runs: K = hi8+0..7 and K = 16+hi8+0..7
// (hi8 = 8 for lanes 16-31). Both runs are 16-byte aligned for our pitches,
// so each fragment is two b128 loads + a shuffle (no per-element packing).
__device__ __forceinline__ v16bf frag_load(const __bf16* p, int hi8) {
  v8bf lo = *(const v8bf*)(p + hi8);
  v8bf hi = *(const v8bf*)(p + 16 + hi8);
  return __builtin_shufflevector(lo, hi, 0, 1, 2, 3, 4, 5, 6, 7,
                                         8, 9, 10, 11, 12, 13, 14, 15);
}
__device__ __forceinline__ v16bf frag_load(const float* p, int hi8) {
  v8f lo = *(const v8f*)(p + hi8);
  v8f hi = *(const v8f*)(p + 16 + hi8);
  v8bf lob = __builtin_convertvector(lo, v8bf);
  v8bf hib = __builtin_convertvector(hi, v8bf);
  return __builtin_shufflevector(lob, hib, 0, 1, 2, 3, 4, 5, 6, 7,
                                           8, 9, 10, 11, 12, 13, 14, 15);
}

__device__ __forceinline__ __bf16 to_bf(float x) { return (__bf16)x; }

// ---------------------------------------------------------------------------
// Generic C[M,N] = A[M,K] * B[N,K]^T, f32 accumulate, bf16 WMMA multiplicands.
// Block: 256 threads = 8 waves laid out 4(M) x 2(N); wave tile = 32x64.
// ---------------------------------------------------------------------------
template <typename AT>
__global__ __launch_bounds__(256)
void gemm_nt_kernel(const AT* __restrict__ A, const float* __restrict__ Bw,
                    float* __restrict__ C, int M, int N, int K) {
  const int lane = threadIdx.x & 31;
  const int wid  = threadIdx.x >> 5;
  const int wm   = wid >> 1;                 // 0..3
  const int wn   = wid & 1;                  // 0..1
  const int m0   = blockIdx.y * 128 + wm * 32;
  const int n0   = blockIdx.x * 128 + wn * 64;
  const int lr   = lane & 15;
  const int hi8  = (lane & 16) >> 1;

  const AT*    arow[2];
  const float* brow[4];
#pragma unroll
  for (int mt = 0; mt < 2; ++mt) arow[mt] = A + (size_t)(m0 + mt * 16 + lr) * K;
#pragma unroll
  for (int nt = 0; nt < 4; ++nt) brow[nt] = Bw + (size_t)(n0 + nt * 16 + lr) * K;

  v8f acc[2][4];
  for (int mt = 0; mt < 2; ++mt)
    for (int nt = 0; nt < 4; ++nt)
      for (int e = 0; e < 8; ++e) acc[mt][nt][e] = 0.0f;

  for (int kk = 0; kk < K; kk += 32) {
    v16bf a[2], b[4];
#pragma unroll
    for (int mt = 0; mt < 2; ++mt) {
      a[mt] = frag_load(arow[mt] + kk, hi8);
      __builtin_prefetch(arow[mt] + kk + 128, 0, 0);
    }
#pragma unroll
    for (int nt = 0; nt < 4; ++nt) {
      b[nt] = frag_load(brow[nt] + kk, hi8);
      __builtin_prefetch(brow[nt] + kk + 128, 0, 0);
    }
#pragma unroll
    for (int mt = 0; mt < 2; ++mt)
#pragma unroll
      for (int nt = 0; nt < 4; ++nt)
        acc[mt][nt] = wmma_bf16(a[mt], b[nt], acc[mt][nt]);
  }

#pragma unroll
  for (int mt = 0; mt < 2; ++mt)
#pragma unroll
    for (int nt = 0; nt < 4; ++nt)
#pragma unroll
      for (int r = 0; r < 8; ++r)
        C[(size_t)(m0 + mt * 16 + r + hi8) * N + (n0 + nt * 16 + lr)] = acc[mt][nt][r];
}

// ---------------------------------------------------------------------------
// Per-(t,h): RMSNorm(q,k), rotary(q,k), v = l0*v + l1*ve. One wave per (t,h).
// Lane owns elements {lane, lane+32, lane+64, lane+96}. Rotary pairs
// (d, d+64): d<32 has freq exp(-(d/31)*ln(1024)); d in [32,64) freq 0.
// Outputs bf16 in [H][T][HD] layout.
// ---------------------------------------------------------------------------
__global__ __launch_bounds__(256)
void qk_norm_rope_v_kernel(const float* __restrict__ qkv,
                           const float* __restrict__ ve,
                           const float* __restrict__ lambdas,
                           __bf16* __restrict__ qb, __bf16* __restrict__ kb,
                           __bf16* __restrict__ vb) {
  const int lane = threadIdx.x & 31;
  const int wid  = threadIdx.x >> 5;
  const int g    = blockIdx.x * 8 + wid;
  const int t    = g >> 3;
  const int h    = g & 7;
  const float eps = 1.1920929e-07f;

  const float f  = __expf(-((float)lane / 31.0f) * 6.9314718056f);
  const float th = (float)t * f;
  const float cs = __cosf(th), sn = __sinf(th);

  const float* base  = qkv + (size_t)t * 3072 + h * HDIM;
  const size_t obase = ((size_t)h * T_LEN + t) * HDIM;

#pragma unroll
  for (int which = 0; which < 2; ++which) {   // 0 = q, 1 = k
    const float* p = base + which * 1024;
    float e0 = p[lane], e1 = p[lane + 32], e2 = p[lane + 64], e3 = p[lane + 96];
    float ss = e0 * e0 + e1 * e1 + e2 * e2 + e3 * e3;
    for (int off = 16; off >= 1; off >>= 1) ss += __shfl_xor(ss, off, 32);
    const float r = rsqrtf(ss * (1.0f / 128.0f) + eps);
    e0 *= r; e1 *= r; e2 *= r; e3 *= r;
    const float y0 = e0 * cs + e2 * sn;
    const float y2 = -e0 * sn + e2 * cs;
    __bf16* o = which ? kb : qb;
    o[obase + lane]      = to_bf(y0);
    o[obase + lane + 32] = to_bf(e1);
    o[obase + lane + 64] = to_bf(y2);
    o[obase + lane + 96] = to_bf(e3);
  }

  const float l0 = lambdas[0], l1 = lambdas[1];
  const float* pv = base + 2048;
  const float* pe = ve + (size_t)t * EDIM + h * HDIM;
#pragma unroll
  for (int j = 0; j < 4; ++j) {
    const int d = lane + 32 * j;
    vb[obase + d] = to_bf(l0 * pv[d] + l1 * pe[d]);
  }
}

// ---------------------------------------------------------------------------
// Flash attention, one block = (128 query rows, 1 head), 8 waves x 16 rows.
// K tile staged row-major (pitch 136); V tile staged TRANSPOSED (Vt[hd][key],
// pitch 40) so P*V B-fragments are two ds_load_b128 instead of 16 u16 loads.
// S via WMMA -> padded LDS buffer -> online softmax (one row per lane,
// causal + doc mask) -> O rescale -> P*V via WMMA.
// ---------------------------------------------------------------------------
#define NKT 32
#define K_PITCH  136   // bf16 per staged K row (128 + 8 pad): 272B, 16B-aligned
#define VT_PITCH 40    // bf16 per staged Vt row (32 keys + 8 pad): 80B, 16B-aligned
#define S_PITCH  36    // f32 per score row: 144B, 16B-aligned, conflict-free

__global__ __launch_bounds__(256)
void flash_attn_kernel(const __bf16* __restrict__ qb, const __bf16* __restrict__ kb,
                       const __bf16* __restrict__ vb, const int* __restrict__ docs,
                       __bf16* __restrict__ attb) {
  __shared__ __bf16 Klds[NKT * K_PITCH];
  __shared__ __bf16 Vt[HDIM * VT_PITCH];
  __shared__ float  Sbuf[8][16 * S_PITCH];
  __shared__ float  s_m[8][16], s_l[8][16], s_scale[8][16];
  __shared__ int    s_docs[NKT];

  const int lane = threadIdx.x & 31;
  const int w    = threadIdx.x >> 5;
  const int lr   = lane & 15;
  const int hi8  = (lane & 16) >> 1;
  const int qt   = blockIdx.x;
  const int h    = blockIdx.y;
  const int qr0  = qt * 128 + w * 16;

  // Q fragments for this wave's 16 rows (K-dim = 128 -> 4 frags)
  v16bf aq[4];
  {
    const __bf16* qrow = qb + ((size_t)h * T_LEN + qr0 + lr) * HDIM;
#pragma unroll
    for (int kk = 0; kk < 4; ++kk) aq[kk] = frag_load(qrow + kk * 32, hi8);
  }

  v8f o[8];
  for (int nt = 0; nt < 8; ++nt)
    for (int e = 0; e < 8; ++e) o[nt][e] = 0.0f;

  if (lane < 16) { s_m[w][lane] = -INFINITY; s_l[w][lane] = 0.0f; }
  const int mydoc = docs[qr0 + lr];

  const int jt_max = qt * 4 + 3;
  for (int jt = 0; jt <= jt_max; ++jt) {
    const int j0 = jt * NKT;
    __syncthreads();
    {
      const unsigned int* ks = (const unsigned int*)(kb + ((size_t)h * T_LEN + j0) * HDIM);
      const __bf16*       vs = vb + ((size_t)h * T_LEN + j0) * HDIM;
      unsigned int* kd = (unsigned int*)Klds;
      for (int idx = threadIdx.x; idx < NKT * 64; idx += 256) {
        const int row = idx >> 6, col = idx & 63;     // 64 dwords per 128-bf16 row
        kd[row * (K_PITCH / 2) + col] = ks[row * 64 + col];
        // V: transpose on the way in. dword idx covers (key=row, hd=2*col,2*col+1)
        const v2bf d = *(const v2bf*)(vs + row * HDIM + 2 * col);
        Vt[(2 * col) * VT_PITCH + row]     = d[0];
        Vt[(2 * col + 1) * VT_PITCH + row] = d[1];
      }
      if (threadIdx.x < NKT) s_docs[threadIdx.x] = docs[j0 + threadIdx.x];
    }
    __syncthreads();

    // S(16x32) = Q(16x128) @ K^T(128x32)
#pragma unroll
    for (int nt = 0; nt < 2; ++nt) {
      v8f s;
      for (int e = 0; e < 8; ++e) s[e] = 0.0f;
#pragma unroll
      for (int kk = 0; kk < 4; ++kk) {
        const v16bf b = frag_load(&Klds[(nt * 16 + lr) * K_PITCH + kk * 32], hi8);
        s = wmma_bf16(aq[kk], b, s);
      }
#pragma unroll
      for (int r = 0; r < 8; ++r)
        Sbuf[w][(r + hi8) * S_PITCH + nt * 16 + lr] = s[r];
    }

    // Online softmax: lane r (0..15) owns query row qr0+r
    if (lane < 16) {
      const int row = lane;
      const int ig  = qr0 + row;
      const float mold = s_m[w][row];
      float rmax = mold;
      for (int c = 0; c < NKT; ++c) {
        const int j = j0 + c;
        const bool ok = (j <= ig) && (s_docs[c] == mydoc);
        const float sv = ok ? Sbuf[w][row * S_PITCH + c] * ATTN_SCALE : -INFINITY;
        rmax = fmaxf(rmax, sv);
      }
      const float mnew  = rmax;
      const float scale = (mnew == -INFINITY) ? 1.0f : __expf(mold - mnew);
      float psum = 0.0f;
      for (int c = 0; c < NKT; ++c) {
        const int j = j0 + c;
        const bool ok = (j <= ig) && (s_docs[c] == mydoc);
        const float sv = ok ? Sbuf[w][row * S_PITCH + c] * ATTN_SCALE : -INFINITY;
        const float p  = (sv == -INFINITY) ? 0.0f : __expf(sv - mnew);
        psum += p;
        Sbuf[w][row * S_PITCH + c] = p;   // P stored back in place
      }
      s_scale[w][row] = scale;
      s_m[w][row]     = mnew;
      s_l[w][row]     = s_l[w][row] * scale + psum;
    }

    // Rescale O by per-row correction factor
    float fr[8];
#pragma unroll
    for (int r = 0; r < 8; ++r) fr[r] = s_scale[w][r + hi8];
#pragma unroll
    for (int nt = 0; nt < 8; ++nt)
#pragma unroll
      for (int r = 0; r < 8; ++r) o[nt][r] *= fr[r];

    // O(16x128) += P(16x32) @ V(32x128)
    const v16bf ap = frag_load(&Sbuf[w][lr * S_PITCH], hi8);
#pragma unroll
    for (int nt = 0; nt < 8; ++nt) {
      const v16bf b = frag_load(&Vt[(nt * 16 + lr) * VT_PITCH], hi8);
      o[nt] = wmma_bf16(ap, b, o[nt]);
    }
  }

  float invl[8];
#pragma unroll
  for (int r = 0; r < 8; ++r) invl[r] = 1.0f / s_l[w][r + hi8];
#pragma unroll
  for (int nt = 0; nt < 8; ++nt)
#pragma unroll
    for (int r = 0; r < 8; ++r)
      attb[(size_t)(qr0 + r + hi8) * EDIM + h * HDIM + nt * 16 + lr] =
          to_bf(o[nt][r] * invl[r]);
}

// ---------------------------------------------------------------------------
extern "C" void kernel_launch(void* const* d_in, const int* in_sizes, int n_in,
                              void* d_out, int out_size, void* d_ws, size_t ws_size,
                              hipStream_t stream) {
  const float* x        = (const float*)d_in[0];
  const float* ve       = (const float*)d_in[1];
  const float* qkv_w    = (const float*)d_in[2];   // (3*H*HD, DIM) row-major
  const float* lambdas  = (const float*)d_in[3];
  const float* c_proj_w = (const float*)d_in[4];   // (DIM, H*HD) row-major
  const int*   docs     = (const int*)d_in[5];
  float*       out      = (float*)d_out;

  char* ws = (char*)d_ws;
  float* qkv = (float*)ws;                                    // 4096*3072 f32 = 48 MB
  size_t off = (size_t)T_LEN * 3072 * sizeof(float);
  __bf16* qb   = (__bf16*)(ws + off); off += (size_t)NH * T_LEN * HDIM * 2;
  __bf16* kb   = (__bf16*)(ws + off); off += (size_t)NH * T_LEN * HDIM * 2;
  __bf16* vb   = (__bf16*)(ws + off); off += (size_t)NH * T_LEN * HDIM * 2;
  __bf16* attb = (__bf16*)(ws + off);                         // 4096*1024 bf16

  // 1) qkv = x @ qkv_w^T   (4096 x 3072, K=1024)
  gemm_nt_kernel<float><<<dim3(3072 / 128, T_LEN / 128), 256, 0, stream>>>(
      x, qkv_w, qkv, T_LEN, 3072, EDIM);

  // 2) RMSNorm + rotary on q,k; v = l0*v + l1*ve; emit bf16 [H][T][HD]
  qk_norm_rope_v_kernel<<<(T_LEN * NH) / 8, 256, 0, stream>>>(
      qkv, ve, lambdas, qb, kb, vb);

  // 3) doc-masked causal flash attention -> attb (T x 1024 bf16)
  flash_attn_kernel<<<dim3(T_LEN / 128, NH), 256, 0, stream>>>(
      qb, kb, vb, docs, attb);

  // 4) out = attb @ c_proj_w^T   (4096 x 1024, K=1024)
  gemm_nt_kernel<__bf16><<<dim3(EDIM / 128, T_LEN / 128), 256, 0, stream>>>(
      attb, c_proj_w, out, T_LEN, EDIM, EDIM);
}
